// FrozenSBERT_25572235280657
// MI455X (gfx1250) — matrix operand order repaired and checked by reading, not compiled, not run on hardware
//
#include <hip/hip_runtime.h>
#include <hip/hip_bf16.h>

// ---------------------------------------------------------------------------
// FrozenSBERT forward for MI455X (gfx1250, wave32, WMMA bf16 16x16x32)
// bf16 end-to-end GEMM operands + async global->LDS staging + LDS double buffer
// B=32 S=512 L=12 H=768 NH=12 DH=64 FF=3072
// ---------------------------------------------------------------------------

#define BB   32
#define SS   512
#define LL   12
#define HH   768
#define NHH  12
#define DHH  64
#define FFF  3072
#define MM   (BB * SS)          // 16384 rows

typedef __attribute__((ext_vector_type(16))) __bf16 v16bf;
typedef __attribute__((ext_vector_type(8)))  float  v8f;
typedef __attribute__((ext_vector_type(4)))  int    i32x4;

#if defined(__has_builtin)
#if __has_builtin(__builtin_amdgcn_global_load_async_to_lds_b128) && \
    __has_builtin(__builtin_amdgcn_s_wait_asynccnt)
#define USE_ASYNC 1
#endif
#endif

// async 16-byte global->LDS copy (falls back to VGPR copy if builtin missing)
__device__ __forceinline__ void copy16(const unsigned short* g, unsigned short* l) {
#ifdef USE_ASYNC
    __builtin_amdgcn_global_load_async_to_lds_b128(
        (__attribute__((address_space(1))) i32x4*)const_cast<unsigned short*>(g),
        (__attribute__((address_space(3))) i32x4*)l, 0, 0);
#else
    *reinterpret_cast<uint4*>(l) = *reinterpret_cast<const uint4*>(g);
#endif
}
__device__ __forceinline__ void wait_async() {
#ifdef USE_ASYNC
    __builtin_amdgcn_s_wait_asynccnt(0);
#endif
}

// float -> bf16 (round to nearest even)
__device__ __forceinline__ unsigned short f2bf(float f) {
    unsigned int u = __float_as_uint(f);
    unsigned int r = u + 0x7FFFu + ((u >> 16) & 1u);
    return (unsigned short)(r >> 16);
}

// A-matrix fragment, 16x32 bf16 tile stored row-major [row][k] in LDS.
// lanes 0-15: rows M=0..15, K 0-15;
// lanes 16-31: same rows, K 16-31.
__device__ __forceinline__ v16bf a_frag(const unsigned short* tile, int m0,
                                        int stride, int k0, int lane) {
    union { v16bf v; unsigned int u[8]; } f;
    const unsigned short* p = tile + (size_t)(m0 + (lane & 15)) * stride + k0
                                   + ((lane & 16) ? 16 : 0);
#pragma unroll
    for (int g = 0; g < 8; ++g) {
        int kk = 2 * g;
        f.u[g] = *(const unsigned int*)(p + kk);
    }
    return f.v;
}

// B-matrix fragment, 32x16 bf16 tile stored TRANSPOSED [n][k] in LDS.
__device__ __forceinline__ v16bf b_frag(const unsigned short* tileT, int n0,
                                        int stride, int k0, int lane) {
    union { v16bf v; unsigned int u[8]; } f;
    const unsigned short* p = tileT + (size_t)(n0 + (lane & 15)) * stride + k0
                                    + ((lane & 16) ? 16 : 0);
#pragma unroll
    for (int g = 0; g < 8; ++g)
        f.u[g] = *(const unsigned int*)(p + 2 * g);
    return f.v;
}

// B-matrix fragment from ROW-MAJOR [k][n] tile (strided u16 gathers)
__device__ __forceinline__ v16bf b_frag_rm(const unsigned short* tile, int n0,
                                           int stride, int lane) {
    union { v16bf v; unsigned int u[8]; } f;
    const unsigned short* p = tile + n0 + (lane & 15)
                                   + ((lane & 16) ? 16 * stride : 0);
#pragma unroll
    for (int g = 0; g < 8; ++g) {
        unsigned int lo = p[(2 * g) * stride];
        unsigned int hi = p[(2 * g + 1) * stride];
        f.u[g] = lo | (hi << 16);
    }
    return f.v;
}

__device__ __forceinline__ v8f wmma_bf16(v16bf a, v16bf b, v8f c) {
    return __builtin_amdgcn_wmma_f32_16x16x32_bf16(false, a, false, b,
                                                   (short)0, c, false, false);
}

// ---------------------------------------------------------------------------
// Weight convert + transpose:  Wt[N][K] (bf16) = W[K][N] (f32)
// ---------------------------------------------------------------------------
__global__ __launch_bounds__(256) void convt_kernel(
    const float* __restrict__ W, unsigned short* __restrict__ Wt, int K, int N)
{
    __shared__ float t[32][33];
    const int kb = blockIdx.y * 32, nb = blockIdx.x * 32;
    const int tid = threadIdx.x;
    for (int i = tid; i < 1024; i += 256) {
        int k = i >> 5, n = i & 31;
        t[k][n] = W[(size_t)(kb + k) * N + nb + n];
    }
    __syncthreads();
    for (int i = tid; i < 1024; i += 256) {
        int n = i >> 5, k = i & 31;
        Wt[(size_t)(nb + n) * K + kb + k] = f2bf(t[k][n]);
    }
}

// ---------------------------------------------------------------------------
// GEMM: out = epilogue(A[M,K](bf16) * Wt[N,K](bf16)^T + bias (+ residual))
// Block tile 128x128, 8 waves, wave tile 32x64, K step 32.
// Async global->LDS staging, LDS double buffer. flags bit0 = exact GELU.
// ---------------------------------------------------------------------------
#define BM 128
#define BN 128
#define BK 32
#define LDT 40            // row stride (elems): 80B, 16B aligned

__global__ __launch_bounds__(256) void gemm_bf16_kernel(
    const unsigned short* __restrict__ A, const unsigned short* __restrict__ Wt,
    const float* __restrict__ bias, const float* __restrict__ residual,
    float* __restrict__ outF, unsigned short* __restrict__ outB,
    int M, int N, int K, int flags)
{
    __shared__ alignas(16) unsigned short As[2][BM * LDT];
    __shared__ alignas(16) unsigned short Bs[2][BN * LDT];

    const int bm = blockIdx.y * BM;
    const int bn = blockIdx.x * BN;
    const int tid = threadIdx.x;
    const int lane = tid & 31;
    const int wid = tid >> 5;
    const int wm = (wid & 3) * 32;
    const int wn = (wid >> 2) * 64;
    const int nk = K / BK;

    v8f acc[2][4];
#pragma unroll
    for (int i = 0; i < 2; ++i)
#pragma unroll
        for (int j = 0; j < 4; ++j) acc[i][j] = (v8f){0,0,0,0,0,0,0,0};

    const unsigned short* Ag = A  + (size_t)bm * K;
    const unsigned short* Wg = Wt + (size_t)bn * K;

    auto stage = [&](int t, int bb) {
        const int k0 = t * BK;
#pragma unroll 2
        for (int i = tid; i < 512; i += 256) {          // 128 rows x 4 chunks
            int r = i >> 2, c = (i & 3) * 8;
            copy16(Ag + (size_t)r * K + k0 + c, &As[bb][r * LDT + c]);
        }
#pragma unroll 2
        for (int i = tid; i < 512; i += 256) {
            int r = i >> 2, c = (i & 3) * 8;
            copy16(Wg + (size_t)r * K + k0 + c, &Bs[bb][r * LDT + c]);
        }
    };

    stage(0, 0);
    for (int t = 0; t < nk; ++t) {
        wait_async();
        __syncthreads();
        if (t + 1 < nk) stage(t + 1, (t + 1) & 1);
        const int bb = t & 1;

        v16bf bf[4];
#pragma unroll
        for (int nt = 0; nt < 4; ++nt)
            bf[nt] = b_frag(Bs[bb], wn + nt * 16, LDT, 0, lane);
#pragma unroll
        for (int mt = 0; mt < 2; ++mt) {
            v16bf af = a_frag(As[bb], wm + mt * 16, LDT, 0, lane);
#pragma unroll
            for (int nt = 0; nt < 4; ++nt)
                acc[mt][nt] = wmma_bf16(af, bf[nt], acc[mt][nt]);
        }
    }

    const int cn = lane & 15;
    const int rb = (lane & 16) ? 8 : 0;
#pragma unroll
    for (int mt = 0; mt < 2; ++mt) {
#pragma unroll
        for (int nt = 0; nt < 4; ++nt) {
            int col = bn + wn + nt * 16 + cn;
            float bv = bias[col];
#pragma unroll
            for (int i = 0; i < 8; ++i) {
                int row = bm + wm + mt * 16 + rb + i;
                float v = acc[mt][nt][i] + bv;
                if (residual) v += residual[(size_t)row * N + col];
                if (flags & 1) v = 0.5f * v * (1.0f + erff(v * 0.70710678118654752f));
                size_t idx = (size_t)row * N + col;
                if (outF) outF[idx] = v;
                if (outB) outB[idx] = f2bf(v);
            }
        }
    }
}

// ---------------------------------------------------------------------------
// Fused attention: one wave per (qtile16, head, batch). bf16 Q/K/V in, bf16 ctx out.
// ---------------------------------------------------------------------------
__global__ __launch_bounds__(32) void attention_kernel(
    const unsigned short* __restrict__ Q, const unsigned short* __restrict__ Kc,
    const unsigned short* __restrict__ V, const float* __restrict__ mask,
    unsigned short* __restrict__ Ctx)
{
    const int qt = blockIdx.x;
    const int hh = blockIdx.y;
    const int b  = blockIdx.z;
    const int lane = threadIdx.x;
    const float scale = 0.125f;          // 1/sqrt(64)

    __shared__ alignas(16) unsigned short Qs[16 * 72];    // [r][d]  (A layout)
    __shared__ alignas(16) unsigned short Ks[16 * 72];    // [key][d] = B^T layout
    __shared__ alignas(16) unsigned short Vsr[32 * 72];   // [key][d] row-major
    __shared__ alignas(16) float          Sc[16 * 520];
    __shared__ alignas(16) unsigned short Ps[16 * 520];

    const size_t rowbase = (size_t)(b * SS) * HH + hh * DHH;

    // Q tile: 16 rows x 8 chunks of 16B
#pragma unroll
    for (int i = lane; i < 128; i += 32) {
        int r = i >> 3, c = (i & 7) * 8;
        copy16(Q + rowbase + (size_t)(qt * 16 + r) * HH + c, &Qs[r * 72 + c]);
    }
    wait_async();
    __syncthreads();

    v16bf qa0 = a_frag(Qs, 0, 72, 0, lane);
    v16bf qa1 = a_frag(Qs, 0, 72, 32, lane);

    const int cn = lane & 15;
    const int rb = (lane & 16) ? 8 : 0;

    // ---- scores over 32 key tiles of 16 ----
    for (int kt = 0; kt < 32; ++kt) {
#pragma unroll
        for (int i = lane; i < 128; i += 32) {
            int r = i >> 3, c = (i & 7) * 8;
            copy16(Kc + rowbase + (size_t)(kt * 16 + r) * HH + c, &Ks[r * 72 + c]);
        }
        wait_async();
        __syncthreads();
        v8f c = (v8f){0,0,0,0,0,0,0,0};
        c = wmma_bf16(qa0, b_frag(Ks, 0, 72, 0, lane), c);
        c = wmma_bf16(qa1, b_frag(Ks, 0, 72, 32, lane), c);
#pragma unroll
        for (int i = 0; i < 8; ++i)
            Sc[(rb + i) * 520 + kt * 16 + cn] = c[i] * scale;
        __syncthreads();
    }

    // ---- masked softmax + renorm: e*m / max(sum(e*m), eps) ----
    const float* mrow = mask + (size_t)b * SS;
    for (int r = 0; r < 16; ++r) {
        float m = -1e30f;
        for (int c = lane; c < SS; c += 32) m = fmaxf(m, Sc[r * 520 + c]);
#pragma unroll
        for (int off = 16; off > 0; off >>= 1)
            m = fmaxf(m, __shfl_xor(m, off, 32));
        float s = 0.f;
        for (int c = lane; c < SS; c += 32) {
            float e = __expf(Sc[r * 520 + c] - m) * mrow[c];
            Sc[r * 520 + c] = e;
            s += e;
        }
#pragma unroll
        for (int off = 16; off > 0; off >>= 1)
            s += __shfl_xor(s, off, 32);
        float inv = 1.0f / fmaxf(s, 1e-9f);
        for (int c = lane; c < SS; c += 32)
            Ps[r * 520 + c] = f2bf(Sc[r * 520 + c] * inv);
    }
    __syncthreads();

    // ---- context: ctx(16x64) = P(16x512) x V(512x64) ----
    v8f acc[4];
#pragma unroll
    for (int nt = 0; nt < 4; ++nt) acc[nt] = (v8f){0,0,0,0,0,0,0,0};

    for (int vt = 0; vt < 16; ++vt) {
#pragma unroll
        for (int i = lane; i < 256; i += 32) {           // 32 rows x 8 chunks
            int r = i >> 3, c = (i & 7) * 8;
            copy16(V + rowbase + (size_t)(vt * 32 + r) * HH + c, &Vsr[r * 72 + c]);
        }
        wait_async();
        __syncthreads();
        v16bf pa = a_frag(Ps, 0, 520, vt * 32, lane);
#pragma unroll
        for (int nt = 0; nt < 4; ++nt)
            acc[nt] = wmma_bf16(pa, b_frag_rm(Vsr, nt * 16, 72, lane), acc[nt]);
        __syncthreads();
    }

#pragma unroll
    for (int nt = 0; nt < 4; ++nt) {
        int d = nt * 16 + cn;
#pragma unroll
        for (int i = 0; i < 8; ++i) {
            int row = qt * 16 + rb + i;
            Ctx[rowbase + (size_t)row * HH + d] = f2bf(acc[nt][i]);
        }
    }
}

// ---------------------------------------------------------------------------
// Row reductions / LN / embed / pool
// ---------------------------------------------------------------------------
__device__ __forceinline__ float block_sum(float v, float* rbuf, int tid) {
    rbuf[tid] = v;
    __syncthreads();
#pragma unroll
    for (int s = 128; s > 0; s >>= 1) {
        if (tid < s) rbuf[tid] += rbuf[tid + s];
        __syncthreads();
    }
    float r = rbuf[0];
    __syncthreads();
    return r;
}

__global__ __launch_bounds__(256) void ln_kernel(
    const float* __restrict__ X, const float* __restrict__ g,
    const float* __restrict__ be, float* __restrict__ Y,
    unsigned short* __restrict__ Yb)
{
    __shared__ float buf[HH];
    __shared__ float rbuf[256];
    const size_t row = blockIdx.x;
    const int tid = threadIdx.x;
    float ls = 0.f;
    for (int j = tid; j < HH; j += 256) { float v = X[row * HH + j]; buf[j] = v; ls += v; }
    float mean = block_sum(ls, rbuf, tid) * (1.0f / HH);
    float lv = 0.f;
    for (int j = tid; j < HH; j += 256) { float d = buf[j] - mean; lv += d * d; }
    float var = block_sum(lv, rbuf, tid) * (1.0f / HH);
    float inv = rsqrtf(var + 1e-12f);
    for (int j = tid; j < HH; j += 256) {
        float o = (buf[j] - mean) * inv * g[j] + be[j];
        if (Y)  Y[row * HH + j] = o;
        if (Yb) Yb[row * HH + j] = f2bf(o);
    }
}

__global__ __launch_bounds__(256) void embed_ln_kernel(
    const int* __restrict__ ids, const float* __restrict__ wemb,
    const float* __restrict__ pemb, const float* __restrict__ temb,
    const float* __restrict__ g, const float* __restrict__ be,
    float* __restrict__ Y, unsigned short* __restrict__ Yb)
{
    __shared__ float buf[HH];
    __shared__ float rbuf[256];
    const size_t row = blockIdx.x;
    const int si = (int)(row % SS);
    const int tid = threadIdx.x;
    const size_t wb = (size_t)ids[row] * HH;
    float ls = 0.f;
    for (int j = tid; j < HH; j += 256) {
        float v = wemb[wb + j] + pemb[(size_t)si * HH + j] + temb[j];
        buf[j] = v; ls += v;
    }
    float mean = block_sum(ls, rbuf, tid) * (1.0f / HH);
    float lv = 0.f;
    for (int j = tid; j < HH; j += 256) { float d = buf[j] - mean; lv += d * d; }
    float var = block_sum(lv, rbuf, tid) * (1.0f / HH);
    float inv = rsqrtf(var + 1e-12f);
    for (int j = tid; j < HH; j += 256) {
        float o = (buf[j] - mean) * inv * g[j] + be[j];
        Y[row * HH + j] = o;
        Yb[row * HH + j] = f2bf(o);
    }
}

__global__ __launch_bounds__(256) void pool_kernel(
    const float* __restrict__ Hf, const float* __restrict__ mask,
    float* __restrict__ out)
{
    __shared__ float buf[HH];
    __shared__ float rbuf[256];
    const int b = blockIdx.x;
    const int tid = threadIdx.x;
    float msum = 0.f;
    for (int s = 0; s < SS; ++s) msum += mask[(size_t)b * SS + s];
    float md = 1.0f / fmaxf(msum, 1e-6f);
    for (int j = tid; j < HH; j += 256) {
        float acc = 0.f;
        for (int s = 0; s < SS; ++s)
            acc += Hf[((size_t)b * SS + s) * HH + j] * mask[(size_t)b * SS + s];
        buf[j] = acc * md;
    }
    __syncthreads();
    float ss = 0.f;
    for (int j = tid; j < HH; j += 256) ss += buf[j] * buf[j];
    float nrm = block_sum(ss, rbuf, tid);
    float scale = 1.0f / fmaxf(sqrtf(nrm), 1e-12f);
    for (int j = tid; j < HH; j += 256)
        out[(size_t)b * HH + j] = buf[j] * scale;
}

// ---------------------------------------------------------------------------
// Host launcher
// ---------------------------------------------------------------------------
extern "C" void kernel_launch(void* const* d_in, const int* in_sizes, int n_in,
                              void* d_out, int out_size, void* d_ws, size_t ws_size,
                              hipStream_t stream)
{
    const int*   ids   = (const int*)  d_in[0];
    const float* amask = (const float*)d_in[1];
    const float* wemb  = (const float*)d_in[2];
    const float* pemb  = (const float*)d_in[3];
    const float* temb  = (const float*)d_in[4];
    const float* lnes  = (const float*)d_in[5];
    const float* lneb  = (const float*)d_in[6];
    const float* Wq    = (const float*)d_in[7];
    const float* bq    = (const float*)d_in[8];
    const float* Wk    = (const float*)d_in[9];
    const float* bk    = (const float*)d_in[10];
    const float* Wv    = (const float*)d_in[11];
    const float* bv    = (const float*)d_in[12];
    const float* Wo    = (const float*)d_in[13];
    const float* bo    = (const float*)d_in[14];
    const float* l1s   = (const float*)d_in[15];
    const float* l1b   = (const float*)d_in[16];
    const float* Wi    = (const float*)d_in[17];
    const float* bi    = (const float*)d_in[18];
    const float* Wf    = (const float*)d_in[19];
    const float* bf_   = (const float*)d_in[20];
    const float* l2s   = (const float*)d_in[21];
    const float* l2b   = (const float*)d_in[22];
    float* out = (float*)d_out;

    // ---- workspace carve-up ----
    char* p = (char*)d_ws;
    auto alloc = [&](size_t bytes) -> char* {
        char* r = p; p += (bytes + 255) & ~(size_t)255; return r;
    };
    const size_t n1 = (size_t)MM * HH;
    float*          hbuf  = (float*)         alloc(n1 * 4);
    float*          tbuf  = (float*)         alloc(n1 * 4);
    float*          attnf = (float*)         alloc(n1 * 4);
    unsigned short* hb    = (unsigned short*)alloc(n1 * 2);
    unsigned short* ab    = (unsigned short*)alloc(n1 * 2);
    unsigned short* qb    = (unsigned short*)alloc(n1 * 2);
    unsigned short* kb2   = (unsigned short*)alloc(n1 * 2);
    unsigned short* vb2   = (unsigned short*)alloc(n1 * 2);
    unsigned short* cb    = (unsigned short*)alloc(n1 * 2);
    unsigned short* ib    = (unsigned short*)alloc((size_t)MM * FFF * 2);
    unsigned short* WqT   = (unsigned short*)alloc((size_t)LL * HH * HH * 2);
    unsigned short* WkT   = (unsigned short*)alloc((size_t)LL * HH * HH * 2);
    unsigned short* WvT   = (unsigned short*)alloc((size_t)LL * HH * HH * 2);
    unsigned short* WoT   = (unsigned short*)alloc((size_t)LL * HH * HH * 2);
    unsigned short* WiT   = (unsigned short*)alloc((size_t)LL * HH * FFF * 2);
    unsigned short* WfT   = (unsigned short*)alloc((size_t)LL * FFF * HH * 2);

    dim3 gchh(HH / 32, HH / 32);
    dim3 gcif(FFF / 32, HH / 32);      // Wi: K=HH rows, N=FFF
    dim3 gcfi(HH / 32, FFF / 32);      // Wf: K=FFF rows, N=HH
    for (int l = 0; l < LL; ++l) {
        convt_kernel<<<gchh, 256, 0, stream>>>(Wq + (size_t)l * HH * HH,  WqT + (size_t)l * HH * HH,  HH,  HH);
        convt_kernel<<<gchh, 256, 0, stream>>>(Wk + (size_t)l * HH * HH,  WkT + (size_t)l * HH * HH,  HH,  HH);
        convt_kernel<<<gchh, 256, 0, stream>>>(Wv + (size_t)l * HH * HH,  WvT + (size_t)l * HH * HH,  HH,  HH);
        convt_kernel<<<gchh, 256, 0, stream>>>(Wo + (size_t)l * HH * HH,  WoT + (size_t)l * HH * HH,  HH,  HH);
        convt_kernel<<<gcif, 256, 0, stream>>>(Wi + (size_t)l * HH * FFF, WiT + (size_t)l * HH * FFF, HH,  FFF);
        convt_kernel<<<gcfi, 256, 0, stream>>>(Wf + (size_t)l * FFF * HH, WfT + (size_t)l * FFF * HH, FFF, HH);
    }

    dim3 g768(HH / BN, MM / BM);       // 6 x 128
    dim3 g3072(FFF / BN, MM / BM);     // 24 x 128
    dim3 gattn(SS / 16, NHH, BB);      // 32 x 12 x 32

    embed_ln_kernel<<<MM, 256, 0, stream>>>(ids, wemb, pemb, temb, lnes, lneb, hbuf, hb);

    for (int l = 0; l < LL; ++l) {
        gemm_bf16_kernel<<<g768, 256, 0, stream>>>(hb, WqT + (size_t)l * HH * HH,
            bq + (size_t)l * HH, nullptr, nullptr, qb, MM, HH, HH, 0);
        gemm_bf16_kernel<<<g768, 256, 0, stream>>>(hb, WkT + (size_t)l * HH * HH,
            bk + (size_t)l * HH, nullptr, nullptr, kb2, MM, HH, HH, 0);
        gemm_bf16_kernel<<<g768, 256, 0, stream>>>(hb, WvT + (size_t)l * HH * HH,
            bv + (size_t)l * HH, nullptr, nullptr, vb2, MM, HH, HH, 0);

        attention_kernel<<<gattn, 32, 0, stream>>>(qb, kb2, vb2, amask, cb);

        gemm_bf16_kernel<<<g768, 256, 0, stream>>>(cb, WoT + (size_t)l * HH * HH,
            bo + (size_t)l * HH, hbuf, tbuf, nullptr, MM, HH, HH, 0);
        ln_kernel<<<MM, 256, 0, stream>>>(tbuf, l1s + (size_t)l * HH,
                                          l1b + (size_t)l * HH, attnf, ab);

        gemm_bf16_kernel<<<g3072, 256, 0, stream>>>(ab, WiT + (size_t)l * HH * FFF,
            bi + (size_t)l * FFF, nullptr, nullptr, ib, MM, FFF, HH, 1);
        gemm_bf16_kernel<<<g768, 256, 0, stream>>>(ib, WfT + (size_t)l * FFF * HH,
            bf_ + (size_t)l * HH, attnf, tbuf, nullptr, MM, HH, FFF, 0);
        ln_kernel<<<MM, 256, 0, stream>>>(tbuf, l2s + (size_t)l * HH,
                                          l2b + (size_t)l * HH, hbuf, hb);
    }

    pool_kernel<<<BB, 256, 0, stream>>>(hbuf, amask, out);
}
